// Self_Attention_15977278341749
// MI455X (gfx1250) — matrix operand order repaired and checked by reading, not compiled, not run on hardware
//
#include <hip/hip_runtime.h>
#include <hip/hip_bf16.h>
#include <stdint.h>

typedef __attribute__((ext_vector_type(16))) _Float16 v16h;
typedef __attribute__((ext_vector_type(8)))  float    v8f;

// ---------------------------------------------------------------------------
// CDNA5 async global->LDS (ASYNCcnt-tracked) helpers, per CDNA5 ISA.
// Each lane copies 16B: LDS[vdst_lane] = MEM[vaddr_lane].
// Low 32 bits of a flat shared-pointer == LDS byte offset (aperture rule).
// ---------------------------------------------------------------------------
__device__ __forceinline__ uint32_t lds_off(const void* p) {
  return (uint32_t)(uintptr_t)p;
}
__device__ __forceinline__ void async_load_b128(uint32_t dst_lds, const void* g) {
  asm volatile("global_load_async_to_lds_b128 %0, %1, off"
               :: "v"(dst_lds), "v"((uint64_t)(uintptr_t)g)
               : "memory");
}
__device__ __forceinline__ void wait_async_le8() {
  asm volatile("s_wait_asynccnt 0x8" ::: "memory");
}
__device__ __forceinline__ void wait_async_0() {
  asm volatile("s_wait_asynccnt 0x0" ::: "memory");
}

// ---------------------------------------------------------------------------
// WMMA fragment helpers (ISA 7.12.2 layouts, wave32)
// ---------------------------------------------------------------------------
__device__ __forceinline__ v16h load_A_frag(const _Float16* base, int ld) {
  const int lane = threadIdx.x & 31;
  const int r    = lane & 15;
  const int koff = (lane & 16) ? 8 : 0;
  const _Float16* p = base + r * ld + koff;
  v16h a;
#pragma unroll
  for (int j = 0; j < 8; ++j) { a[j] = p[j]; a[8 + j] = p[16 + j]; }
  return a;
}

__device__ __forceinline__ v16h load_B_fragT(const _Float16* base, int ld) {
  // logical B[k][n] stored n-major: element = base[n*ld + k]
  const int lane  = threadIdx.x & 31;
  const int n     = lane & 15;
  const int kbase = (lane & 16) ? 8 : 0;
  const _Float16* p = base + n * ld + kbase;
  v16h b;
#pragma unroll
  for (int j = 0; j < 8; ++j) { b[j] = p[j]; b[8 + j] = p[16 + j]; }
  return b;
}

__device__ __forceinline__ v8f wmma_f16(v16h a, v16h b, v8f c) {
  return __builtin_amdgcn_wmma_f32_16x16x32_f16(false, a, false, b, (short)0, c,
                                                false, false);
}

// ---------------------------------------------------------------------------
// Kernel 1: QKV projection.  X[16384,256](f32) x W[256,256](f32) -> f16.
// grid = (256, 3), block = 256 (8 waves). Wave w: rows (w&3)*16, cols (w>>2)*128.
// Epilogue stages the f16 tile in LDS and writes coalesced b128.
// ---------------------------------------------------------------------------
#define X_LD 36    // 32+4 : A-frag stride 18 DW, gcd(18,64)=2
#define W_LD 34    // 32+2 : B-frag stride 17 DW, gcd(17,64)=1
#define ST_LD 264  // 256+8: 528B rows, 16B aligned for b128 staging reads

__global__ __launch_bounds__(256) void qkv_proj_kernel(
    const float* __restrict__ x, const float* __restrict__ w3,
    _Float16* __restrict__ qkv) {
  __shared__ union {
    struct { _Float16 X[64 * X_LD]; _Float16 Wt[256 * W_LD]; } t;
    _Float16 stage[64 * ST_LD];
  } sm;

  const int tid     = threadIdx.x;
  const int wave    = tid >> 5;
  const int lane    = tid & 31;
  const int rowTile = blockIdx.x * 64;
  const int which   = blockIdx.y;

  const float* W   = w3 + (size_t)which * 256 * 256;
  _Float16*    out = qkv + (size_t)which * 16384 * 256;

  const int wrow = (wave & 3) * 16;
  const int wcol = (wave >> 2) * 128;

  v8f acc[8] = {};

  for (int k0 = 0; k0 < 256; k0 += 32) {
    for (int i = 0; i < 8; ++i) {               // X tile 64x32 (f32->f16)
      int idx = tid + i * 256;
      int r = idx >> 5, c = idx & 31;
      sm.t.X[r * X_LD + c] = (_Float16)x[(size_t)(rowTile + r) * 256 + k0 + c];
    }
    for (int i = 0; i < 32; ++i) {              // W tile 32x256, n-major
      sm.t.Wt[tid * W_LD + i] = (_Float16)W[(size_t)(k0 + i) * 256 + tid];
    }
    __syncthreads();

    v16h a = load_A_frag(sm.t.X + wrow * X_LD, X_LD);
#pragma unroll
    for (int t = 0; t < 8; ++t) {
      v16h b = load_B_fragT(sm.t.Wt + (wcol + t * 16) * W_LD, W_LD);
      acc[t] = wmma_f16(a, b, acc[t]);
    }
    __syncthreads();
  }

  // stage f16 tile in LDS, then coalesced b128 stores
  const int nn      = lane & 15;
  const int halfoff = (lane & 16) ? 8 : 0;
#pragma unroll
  for (int t = 0; t < 8; ++t)
#pragma unroll
    for (int r = 0; r < 8; ++r)
      sm.stage[(wrow + halfoff + r) * ST_LD + wcol + t * 16 + nn] =
          (_Float16)acc[t][r];
  __syncthreads();

  for (int i = 0; i < 8; ++i) {
    int c = tid + i * 256;                      // 2048 chunks of 8 halves
    int row = c >> 5, col8 = (c & 31) * 8;
    uint4 w = *(const uint4*)&sm.stage[row * ST_LD + col8];
    *(uint4*)&out[(size_t)(rowTile + row) * 256 + col8] = w;
  }
}

// ---------------------------------------------------------------------------
// Kernel 2: flash attention.  grid = (32, 8), block = 256 (8 waves).
// Double-buffered async K stream; Q async-loaded once; V transposed via VGPRs.
// ---------------------------------------------------------------------------
#define Q_LD 264   // 528B rows: 16B-aligned async dest, gcd(132,64)=4
#define K_LD 264
#define VT_LD 66   // 132B stride: gcd(33,64)=1 -> conflict-free frag reads
#define P_LD 68    // 136B stride: gcd(34,64)=2

__global__ __launch_bounds__(256) void flash_attn_kernel(
    const _Float16* __restrict__ qkv, const unsigned char* __restrict__ mask,
    float* __restrict__ out) {
  constexpr int   S = 2048, D = 256, NKB = S / 64;
  constexpr float SCALE = 0.125f;  // 1/sqrt(64), hardcoded in reference

  __shared__ _Float16 sQ[64 * Q_LD];
  __shared__ _Float16 sK[2][64 * K_LD];
  __shared__ _Float16 sVt[256 * VT_LD];  // [d][key]
  __shared__ float    sS[64 * 64];
  __shared__ _Float16 sP[64 * P_LD];
  __shared__ float    sAlpha[64];
  __shared__ float    sM[64];
  __shared__ float    sL[64];

  const int tid   = threadIdx.x;
  const int lane  = tid & 31;
  const int wave  = tid >> 5;
  const int bb    = blockIdx.y;
  const int qBase = blockIdx.x * 64;

  const _Float16* Q = qkv + (size_t)bb * S * D;
  const _Float16* K = qkv + (size_t)1 * 16384 * 256 + (size_t)bb * S * D;
  const _Float16* V = qkv + (size_t)2 * 16384 * 256 + (size_t)bb * S * D;

  const uint32_t sQoff  = lds_off(sQ);
  const uint32_t sKoff0 = lds_off(sK[0]);
  const uint32_t sKoff1 = lds_off(sK[1]);

  // async Q tile (64x256 f16 = 2048 16B chunks, 8 per thread)
#pragma unroll
  for (int i = 0; i < 8; ++i) {
    int c = tid + i * 256, row = c >> 5, col8 = (c & 31) * 8;
    async_load_b128(sQoff + (uint32_t)(row * Q_LD + col8) * 2,
                    Q + (size_t)(qBase + row) * D + col8);
  }
  // async prefetch K block 0
#pragma unroll
  for (int i = 0; i < 8; ++i) {
    int c = tid + i * 256, row = c >> 5, col8 = (c & 31) * 8;
    async_load_b128(sKoff0 + (uint32_t)(row * K_LD + col8) * 2,
                    K + (size_t)row * D + col8);
  }
  if (tid < 64) { sM[tid] = -1e30f; sL[tid] = 0.0f; }

  const int rowblk  = (wave & 3) * 16;
  const int scol    = (wave >> 2) * 32;   // score phase: 2 col tiles
  const int ocol    = (wave >> 2) * 128;  // output phase: 8 col tiles
  const int halfoff = (lane & 16) ? 8 : 0;
  const int nn      = lane & 15;

  v8f oacc[8] = {};

  for (int kb = 0; kb < NKB; ++kb) {
    const int      cur     = kb & 1;
    const uint32_t nxtOff  = (cur ? sKoff0 : sKoff1);
    const _Float16* sKcur  = sK[cur];

    __syncthreads();  // previous compute done: sK[nxt] and sVt reusable

    // async prefetch of K(kb+1) overlaps this iteration's compute
    if (kb + 1 < NKB) {
#pragma unroll
      for (int i = 0; i < 8; ++i) {
        int c = tid + i * 256, row = c >> 5, col8 = (c & 31) * 8;
        async_load_b128(nxtOff + (uint32_t)(row * K_LD + col8) * 2,
                        K + (size_t)((kb + 1) * 64 + row) * D + col8);
      }
    }
    // V tile -> transposed LDS (b128 global loads, b16 LDS scatter)
    for (int i = 0; i < 8; ++i) {
      int c = tid + i * 256, row = c >> 5, col8 = (c & 31) * 8;
      uint4 w = *(const uint4*)(V + (size_t)(kb * 64 + row) * D + col8);
      const _Float16* hv = (const _Float16*)&w;
#pragma unroll
      for (int j = 0; j < 8; ++j) sVt[(col8 + j) * VT_LD + row] = hv[j];
    }
    if (kb + 1 < NKB) wait_async_le8();  // K(kb) arrived; K(kb+1) in flight
    else             wait_async_0();
    __syncthreads();

    // ---- Sc = Q * K^T  (64x64), each wave 2 tiles -------------------------
    v8f sc0 = {}, sc1 = {};
#pragma unroll
    for (int k0 = 0; k0 < D; k0 += 32) {
      v16h a  = load_A_frag(sQ + rowblk * Q_LD + k0, Q_LD);
      v16h b0 = load_B_fragT(sKcur + (scol)      * K_LD + k0, K_LD);
      v16h b1 = load_B_fragT(sKcur + (scol + 16) * K_LD + k0, K_LD);
      sc0 = wmma_f16(a, b0, sc0);
      sc1 = wmma_f16(a, b1, sc1);
    }
#pragma unroll
    for (int r = 0; r < 8; ++r) {
      sS[(rowblk + halfoff + r) * 64 + scol + nn]      = sc0[r] * SCALE;
      sS[(rowblk + halfoff + r) * 64 + scol + 16 + nn] = sc1[r] * SCALE;
    }
    __syncthreads();

    // ---- online softmax: one thread per query row ------------------------
    if (tid < 64) {
      const int   row  = tid;
      const float mval = mask[(size_t)bb * S + qBase + row] ? 1.0f : 0.0f;
      float mOld = sM[row];
      float mNew = mOld;
      for (int j = 0; j < 64; ++j)
        mNew = fmaxf(mNew, sS[row * 64 + j] * mval);
      float alpha = __expf(mOld - mNew);
      float lsum  = 0.0f;
      for (int j = 0; j < 64; ++j) {
        float p = __expf(sS[row * 64 + j] * mval - mNew);
        lsum += p;
        sP[row * P_LD + j] = (_Float16)p;
      }
      sM[row]     = mNew;
      sL[row]     = sL[row] * alpha + lsum;
      sAlpha[row] = alpha;
    }
    __syncthreads();

    // ---- rescale O accumulators, then O += P * V -------------------------
    float al[8];
#pragma unroll
    for (int r = 0; r < 8; ++r) al[r] = sAlpha[rowblk + halfoff + r];
#pragma unroll
    for (int t = 0; t < 8; ++t)
#pragma unroll
      for (int r = 0; r < 8; ++r) oacc[t][r] *= al[r];

#pragma unroll
    for (int k0 = 0; k0 < 64; k0 += 32) {
      v16h a = load_A_frag(sP + rowblk * P_LD + k0, P_LD);
#pragma unroll
      for (int t = 0; t < 8; ++t) {
        v16h b = load_B_fragT(sVt + (ocol + t * 16) * VT_LD + k0, VT_LD);
        oacc[t] = wmma_f16(a, b, oacc[t]);
      }
    }
  }
  __syncthreads();

  float inv[8];
#pragma unroll
  for (int r = 0; r < 8; ++r) inv[r] = 1.0f / sL[rowblk + halfoff + r];
#pragma unroll
  for (int t = 0; t < 8; ++t)
#pragma unroll
    for (int r = 0; r < 8; ++r) {
      int row = qBase + rowblk + halfoff + r;
      int col = ocol + t * 16 + nn;
      out[((size_t)bb * S + row) * D + col] = oacc[t][r] * inv[r];
    }
}

// ---------------------------------------------------------------------------
extern "C" void kernel_launch(void* const* d_in, const int* in_sizes, int n_in,
                              void* d_out, int out_size, void* d_ws, size_t ws_size,
                              hipStream_t stream) {
  (void)in_sizes; (void)n_in; (void)out_size; (void)ws_size;
  const float*         x    = (const float*)d_in[0];          // [8,2048,256] f32
  const float*         kern = (const float*)d_in[1];          // [3,256,256]  f32
  const unsigned char* mask = (const unsigned char*)d_in[2];  // [8,2048] bool (1 byte)
  float*               out  = (float*)d_out;                  // [8,2048,256] f32
  _Float16*            qkv  = (_Float16*)d_ws;                // 3 * 16384*256 f16 (~24 MB)

  dim3 gProj(16384 / 64, 3);
  qkv_proj_kernel<<<gProj, 256, 0, stream>>>(x, kern, qkv);

  dim3 gAttn(2048 / 64, 8);
  flash_attn_kernel<<<gAttn, 256, 0, stream>>>(qkv, mask, out);
}
